// GATLayer_11467562680537
// MI455X (gfx1250) — matrix operand order repaired and checked by reading, not compiled, not run on hardware
//
#include <hip/hip_runtime.h>
#include <hip/hip_bf16.h>
#include <cstdint>
#include <cstddef>

#define N_NODES 50000
#define N_EDGES 1000000
#define D_FEAT  512
#define F_ATTN  64
#define SLOPE   0.2f
#define N_TILES (N_NODES / 16)   // 3125 exactly

typedef __attribute__((ext_vector_type(16))) __bf16 v16bf;
typedef __attribute__((ext_vector_type(8)))  float  v8f;
typedef __attribute__((ext_vector_type(4)))  float  f32x4;

// Monotonic int key for float atomicMax (order-preserving bijection).
__device__ __forceinline__ int fkey(float f) {
    int i = __float_as_int(f);
    return (i >= 0) ? i : (i ^ 0x7fffffff);
}
__device__ __forceinline__ float funkey(int k) {
    return __int_as_float((k >= 0) ? k : (k ^ 0x7fffffff));
}

// ---------------------------------------------------------------------------
// 0) init: zero output accumulator, reset segment-max keys and denominators
// ---------------------------------------------------------------------------
__global__ void init_kernel(float* __restrict__ out,
                            int* __restrict__ emax_key,
                            float* __restrict__ denom) {
    int i = blockIdx.x * blockDim.x + threadIdx.x;
    if (i < N_NODES * F_ATTN) out[i] = 0.0f;
    if (i < N_NODES) {
        emax_key[i] = (int)0x80000000u;  // key(-inf)
        denom[i]    = 0.0f;
    }
}

// ---------------------------------------------------------------------------
// 1) Projection: z = select(node_type, d_sim@Wd.T, m_sim@Wm.T) via bf16 WMMA.
//    8 waves/block, each wave owns one 16-row tile; Wd/Wm staged in LDS (bf16).
// ---------------------------------------------------------------------------
__launch_bounds__(256, 1)
__global__ void proj_kernel(const float* __restrict__ d_sim,
                            const float* __restrict__ m_sim,
                            const float* __restrict__ Wd,
                            const float* __restrict__ Wm,
                            const int*   __restrict__ node_type,
                            float* __restrict__ z) {
    // 2 x 64 x 512 bf16 = 128 KB of the WGP's 320 KB LDS
    __shared__ __align__(32) __bf16 sW[2][F_ATTN][D_FEAT];

    const int tid = threadIdx.x;
    for (int i = tid; i < F_ATTN * D_FEAT; i += 256) {
        ((__bf16*)sW[0])[i] = (__bf16)Wd[i];
        ((__bf16*)sW[1])[i] = (__bf16)Wm[i];
    }
    __syncthreads();

    const int lane = tid & 31;
    const int wave = tid >> 5;
    const int tile = blockIdx.x * 8 + wave;
    if (tile >= N_TILES) return;

    const int hl  = lane >> 4;          // which half of the wave
    const int l15 = lane & 15;
    const int row = tile * 16 + l15;    // A-matrix row owned by this lane
    const int kb  = hl * 16;            // A-frag K sub-offset (contiguous K layout)

    v8f acc_d[4], acc_m[4];
#pragma unroll
    for (int nt = 0; nt < 4; ++nt) { acc_d[nt] = (v8f)0.0f; acc_m[nt] = (v8f)0.0f; }

    for (int k0 = 0; k0 < D_FEAT; k0 += 32) {
        const size_t abase = (size_t)row * D_FEAT + (size_t)(k0 + kb);
        // A-frag: lane holds K = kb..kb+15 (elems 0..15)
        f32x4 da0 = *(const f32x4*)(d_sim + abase);
        f32x4 da1 = *(const f32x4*)(d_sim + abase + 4);
        f32x4 da2 = *(const f32x4*)(d_sim + abase + 8);
        f32x4 da3 = *(const f32x4*)(d_sim + abase + 12);
        f32x4 ma0 = *(const f32x4*)(m_sim + abase);
        f32x4 ma1 = *(const f32x4*)(m_sim + abase + 4);
        f32x4 ma2 = *(const f32x4*)(m_sim + abase + 8);
        f32x4 ma3 = *(const f32x4*)(m_sim + abase + 12);

        v16bf a_d, a_m;
#pragma unroll
        for (int i = 0; i < 4; ++i) {
            a_d[i]      = (__bf16)da0[i];
            a_d[4 + i]  = (__bf16)da1[i];
            a_d[8 + i]  = (__bf16)da2[i];
            a_d[12 + i] = (__bf16)da3[i];
            a_m[i]      = (__bf16)ma0[i];
            a_m[4 + i]  = (__bf16)ma1[i];
            a_m[8 + i]  = (__bf16)ma2[i];
            a_m[12 + i] = (__bf16)ma3[i];
        }

        // B-frag: lane owns column n = nt*16 + l15, contiguous K range of 16
        const int bk = k0 + hl * 16;
#pragma unroll
        for (int nt = 0; nt < 4; ++nt) {
            const v16bf b_d = *(const v16bf*)&sW[0][nt * 16 + l15][bk];
            const v16bf b_m = *(const v16bf*)&sW[1][nt * 16 + l15][bk];
            acc_d[nt] = __builtin_amdgcn_wmma_f32_16x16x32_bf16(
                false, a_d, false, b_d, (short)0, acc_d[nt], false, false);
            acc_m[nt] = __builtin_amdgcn_wmma_f32_16x16x32_bf16(
                false, a_m, false, b_m, (short)0, acc_m[nt], false, false);
        }
    }

    // Epilogue: C/D layout — lane holds N=l15, VGPR r holds M = r + 8*hl
#pragma unroll
    for (int r = 0; r < 8; ++r) {
        const int m = tile * 16 + r + 8 * hl;
        const int t = node_type[m];
#pragma unroll
        for (int nt = 0; nt < 4; ++nt) {
            float v = (t == 1) ? acc_d[nt][r] : acc_m[nt][r];
            z[(size_t)m * F_ATTN + nt * 16 + l15] = v;
        }
    }
}

// ---------------------------------------------------------------------------
// 2) Per-node attention scalars: s1 = z . Wa[:64], s2 = z . Wa[64:]
// ---------------------------------------------------------------------------
__global__ void s12_kernel(const float* __restrict__ z,
                           const float* __restrict__ Wa,
                           float* __restrict__ s1,
                           float* __restrict__ s2) {
    const int lane = threadIdx.x & 31;
    const int wave = threadIdx.x >> 5;
    const int n = blockIdx.x * 8 + wave;
    if (n >= N_NODES) return;
    const float z0 = z[(size_t)n * F_ATTN + lane];
    const float z1 = z[(size_t)n * F_ATTN + 32 + lane];
    float p1 = z0 * Wa[lane]      + z1 * Wa[32 + lane];
    float p2 = z0 * Wa[64 + lane] + z1 * Wa[96 + lane];
#pragma unroll
    for (int o = 16; o > 0; o >>= 1) {
        p1 += __shfl_xor(p1, o);
        p2 += __shfl_xor(p2, o);
    }
    if (lane == 0) { s1[n] = p1; s2[n] = p2; }
}

// ---------------------------------------------------------------------------
// 3) Edge logits + segment max (int-punned float atomicMax)
// ---------------------------------------------------------------------------
__global__ void edge_logit_kernel(const int* __restrict__ esrc,
                                  const int* __restrict__ edst,
                                  const float* __restrict__ s1,
                                  const float* __restrict__ s2,
                                  float* __restrict__ e,
                                  int* __restrict__ emax_key) {
    int j = blockIdx.x * blockDim.x + threadIdx.x;
    if (j >= N_EDGES) return;
    const int d = edst[j];
    float v = s1[esrc[j]] + s2[d];
    v = (v > 0.0f) ? v : SLOPE * v;  // leaky relu
    e[j] = v;
    atomicMax(&emax_key[d], fkey(v));
}

// ---------------------------------------------------------------------------
// 4) exp(e - max) and segment sum of denominators (overwrites e with ex)
// ---------------------------------------------------------------------------
__global__ void edge_exp_kernel(const int* __restrict__ edst,
                                float* __restrict__ e,
                                const int* __restrict__ emax_key,
                                float* __restrict__ denom) {
    int j = blockIdx.x * blockDim.x + threadIdx.x;
    if (j >= N_EDGES) return;
    const int d = edst[j];
    const float ex = __expf(e[j] - funkey(emax_key[d]));
    e[j] = ex;
    atomicAdd(&denom[d], ex);
}

// ---------------------------------------------------------------------------
// 5) Weighted scatter: out[dst] += alpha * z[src]   (one thread per edge-feat)
// ---------------------------------------------------------------------------
__global__ void edge_scatter_kernel(const int* __restrict__ esrc,
                                    const int* __restrict__ edst,
                                    const float* __restrict__ ex,
                                    const float* __restrict__ denom,
                                    const float* __restrict__ z,
                                    float* __restrict__ out) {
    long long idx = (long long)blockIdx.x * blockDim.x + threadIdx.x;
    if (idx >= (long long)N_EDGES * F_ATTN) return;
    const int j = (int)(idx >> 6);
    const int f = (int)(idx & (F_ATTN - 1));
    const int d = edst[j];
    const float alpha = ex[j] / denom[d];
    atomicAdd(&out[(size_t)d * F_ATTN + f],
              alpha * z[(size_t)esrc[j] * F_ATTN + f]);
}

// ---------------------------------------------------------------------------
// 6) elu in place
// ---------------------------------------------------------------------------
__global__ void elu_kernel(float* __restrict__ out) {
    int i = blockIdx.x * blockDim.x + threadIdx.x;
    if (i >= N_NODES * F_ATTN) return;
    const float v = out[i];
    out[i] = (v > 0.0f) ? v : (__expf(v) - 1.0f);
}

// ---------------------------------------------------------------------------
extern "C" void kernel_launch(void* const* d_in, const int* in_sizes, int n_in,
                              void* d_out, int out_size, void* d_ws, size_t ws_size,
                              hipStream_t stream) {
    const float* d_sim     = (const float*)d_in[0];
    const float* m_sim     = (const float*)d_in[1];
    const float* Wd        = (const float*)d_in[2];
    const float* Wm        = (const float*)d_in[3];
    const float* Wa        = (const float*)d_in[4];
    const int*   node_type = (const int*)d_in[5];
    const int*   edge_src  = (const int*)d_in[6];
    const int*   edge_dst  = (const int*)d_in[7];
    float*       out       = (float*)d_out;

    // workspace carve-up (all 4-byte types; ~17.6 MB total)
    char* ws = (char*)d_ws;
    float* z        = (float*)ws;                         ws += (size_t)N_NODES * F_ATTN * 4;
    float* s1       = (float*)ws;                         ws += (size_t)N_NODES * 4;
    float* s2       = (float*)ws;                         ws += (size_t)N_NODES * 4;
    float* e        = (float*)ws;                         ws += (size_t)N_EDGES * 4;
    int*   emax_key = (int*)ws;                           ws += (size_t)N_NODES * 4;
    float* denom    = (float*)ws;

    const int NF = N_NODES * F_ATTN;

    init_kernel<<<(NF + 255) / 256, 256, 0, stream>>>(out, emax_key, denom);

    proj_kernel<<<(N_TILES + 7) / 8, 256, 0, stream>>>(
        d_sim, m_sim, Wd, Wm, node_type, z);

    s12_kernel<<<(N_NODES + 7) / 8, 256, 0, stream>>>(z, Wa, s1, s2);

    edge_logit_kernel<<<(N_EDGES + 255) / 256, 256, 0, stream>>>(
        edge_src, edge_dst, s1, s2, e, emax_key);

    edge_exp_kernel<<<(N_EDGES + 255) / 256, 256, 0, stream>>>(
        edge_dst, e, emax_key, denom);

    {
        long long total = (long long)N_EDGES * F_ATTN;
        int blocks = (int)((total + 255) / 256);
        edge_scatter_kernel<<<blocks, 256, 0, stream>>>(
            edge_src, edge_dst, e, denom, z, out);
    }

    elu_kernel<<<(NF + 255) / 256, 256, 0, stream>>>(out);
}
